// MultiHeadAttention_41747082117805
// MI455X (gfx1250) — matrix operand order repaired and checked by reading, not compiled, not run on hardware
//
#include <hip/hip_runtime.h>
#include <hip/hip_bf16.h>

typedef unsigned short ushort_t;
typedef unsigned int uint_t;
typedef __attribute__((ext_vector_type(16))) __bf16 v16bf;
typedef __attribute__((ext_vector_type(8)))  float  v8f;

// GCC-style vector type matching the async-LDS builtin's expected pointee
typedef int v4i_t __attribute__((vector_size(16)));
typedef __attribute__((address_space(1))) v4i_t* gptr_v4i;   // global
typedef __attribute__((address_space(3))) v4i_t* lptr_v4i;   // LDS

union FragBF { v16bf v; uint_t u[8]; ushort_t s[16]; };

#if defined(__AMDGCN__) && __has_builtin(__builtin_amdgcn_global_load_async_to_lds_b128)
#define HAVE_ASYNC_LDS 1
#else
#define HAVE_ASYNC_LDS 0
#endif

// ---------------------------------------------------------------- helpers
__device__ __forceinline__ ushort_t f32_to_bf16(float f) {
  union { float f; uint_t u; } x; x.f = f;
  uint_t r = x.u + 0x7FFFu + ((x.u >> 16) & 1u);   // round-to-nearest-even
  return (ushort_t)(r >> 16);
}

__device__ __forceinline__ v8f wmma_bf16(const FragBF& a, const FragBF& b, v8f c) {
  // (neg_a, A, neg_b, B, c_mod, C, reuse_a, reuse_b)
  return __builtin_amdgcn_wmma_f32_16x16x32_bf16(false, a.v, false, b.v,
                                                 (short)0, c, false, false);
}

__device__ __forceinline__ void wait_async_lds() {
#if defined(__AMDGCN__)
#if __has_builtin(__builtin_amdgcn_s_wait_asynccnt)
  __builtin_amdgcn_s_wait_asynccnt(0);
#else
  asm volatile("s_wait_asynccnt 0x0" ::: "memory");
#endif
#endif
}

// A-fragment K index for register v, lane-group g (ISA 16-bit A 16x32 layout)
__device__ __forceinline__ int a_kidx(int v, int g) {
  return (v < 4 ? 2 * v : 16 + 2 * (v - 4)) + 8 * g;
}
// B-fragment K index (pair base) for register v, lane-group g (same K order as A)
__device__ __forceinline__ int b_kidx(int v, int g) { return (v < 4 ? 2 * v : 16 + 2 * (v - 4)) + 8 * g; }

// ---------------------------------------------------------------- cast f32->bf16
__global__ void cast_f32_to_bf16(const float* __restrict__ in,
                                 ushort_t* __restrict__ out, int n) {
  int i = blockIdx.x * blockDim.x + threadIdx.x;
  if (i < n) out[i] = f32_to_bf16(in[i]);
}

// ---------------------------------------------------------------- transpose+cast
// in: [K][N] fp32 row-major ; out: [N][K] bf16 row-major. Block (32,8), 32x32 tile.
__global__ void transpose_cast_f32_bf16(const float* __restrict__ in,
                                        ushort_t* __restrict__ out,
                                        int K, int N) {
  __shared__ float tile[32][33];
  const int n0 = blockIdx.x * 32, k0 = blockIdx.y * 32;
  const int tx = threadIdx.x, ty = threadIdx.y;
#pragma unroll
  for (int j = 0; j < 4; ++j) {
    int k = ty + j * 8;
    tile[k][tx] = in[(size_t)(k0 + k) * N + n0 + tx];
  }
  __syncthreads();
#pragma unroll
  for (int j = 0; j < 4; ++j) {
    int n = ty + j * 8;
    out[(size_t)(n0 + n) * K + k0 + tx] = f32_to_bf16(tile[tx][n]);
  }
}

// ---------------------------------------------------------------- generic bf16 GEMM
// C[M,N] = A[M,K] @ Bt[N,K]^T + bias ;  A row-major bf16, Bt = B^T row-major bf16.
// Block tile 128x128, BK=32, 8 waves (4 along M x 2 along N), wave tile 32x64.
template <bool OUT_BF16>
__global__ void __launch_bounds__(256)
gemm_bf16_wmma(const ushort_t* __restrict__ A,
               const ushort_t* __restrict__ Bt,
               const float* __restrict__ bias,
               void* __restrict__ Cout,
               int M, int N, int K) {
  __shared__ __align__(16) ushort_t As[128][32];   // [m][k]
  __shared__ __align__(16) ushort_t Bs[128][32];   // [n][k]

  const int tid = threadIdx.x;
  const int lane = tid & 31, w = tid >> 5;
  const int wm = w >> 1, wn = w & 1;               // 4 x 2 wave grid
  const int g = lane >> 4, r16 = lane & 15;
  const int m0 = blockIdx.y * 128, n0 = blockIdx.x * 128;

  v8f acc[2][4];
#pragma unroll
  for (int a = 0; a < 2; ++a)
#pragma unroll
    for (int s = 0; s < 4; ++s) acc[a][s] = (v8f){};

  for (int k0 = 0; k0 < K; k0 += 32) {
    // ---- stage 128x32 A tile and 128x32 B tile (both K-contiguous, 16B/lane) ----
#if HAVE_ASYNC_LDS
#pragma unroll
    for (int j = 0; j < 2; ++j) {
      int idx = tid + j * 256;          // 0..511
      int row = idx >> 2;               // 0..127
      int cg = (idx & 3) * 8;           // k group (8 bf16 = 16B)
      __builtin_amdgcn_global_load_async_to_lds_b128(
          (gptr_v4i)(A + (size_t)(m0 + row) * K + k0 + cg),
          (lptr_v4i)&As[row][cg], 0, 0);
      __builtin_amdgcn_global_load_async_to_lds_b128(
          (gptr_v4i)(Bt + (size_t)(n0 + row) * K + k0 + cg),
          (lptr_v4i)&Bs[row][cg], 0, 0);
    }
    wait_async_lds();
    __syncthreads();
#else
#pragma unroll
    for (int j = 0; j < 2; ++j) {
      int idx = tid + j * 256;
      int row = idx >> 2;
      int cg = (idx & 3) * 8;
      *(uint4*)&As[row][cg] = *(const uint4*)(A + (size_t)(m0 + row) * K + k0 + cg);
      *(uint4*)&Bs[row][cg] = *(const uint4*)(Bt + (size_t)(n0 + row) * K + k0 + cg);
    }
    __syncthreads();
#endif

    // ---- compute: 2 A-frags x 4 B-frags = 8 WMMAs per wave per K-step ----
    FragBF af[2];
#pragma unroll
    for (int a = 0; a < 2; ++a)
#pragma unroll
      for (int v = 0; v < 8; ++v)
        af[a].u[v] = *(const uint_t*)&As[wm * 32 + a * 16 + r16][a_kidx(v, g)];

#pragma unroll
    for (int s = 0; s < 4; ++s) {
      FragBF bf;
#pragma unroll
      for (int v = 0; v < 8; ++v)
        bf.u[v] = *(const uint_t*)&Bs[wn * 64 + s * 16 + r16][b_kidx(v, g)];
      acc[0][s] = wmma_bf16(af[0], bf, acc[0][s]);
      acc[1][s] = wmma_bf16(af[1], bf, acc[1][s]);
    }
    __syncthreads();
  }

  // ---- store 32x64 wave tile ----
#pragma unroll
  for (int a = 0; a < 2; ++a) {
#pragma unroll
    for (int s = 0; s < 4; ++s) {
#pragma unroll
      for (int r = 0; r < 8; ++r) {
        int m = m0 + wm * 32 + a * 16 + 8 * g + r;
        int n = n0 + wn * 64 + s * 16 + r16;
        float x = acc[a][s][r] + (bias ? bias[n] : 0.f);
        if (OUT_BF16) {
          ((ushort_t*)Cout)[(size_t)m * N + n] = f32_to_bf16(x);
        } else {
          ((float*)Cout)[(size_t)m * N + n] = x;
        }
      }
    }
  }
}

// ---------------------------------------------------------------- Pass A: G_c = K_c^T V_c
// grid (64 chunks, 16 heads), 256 threads.
__global__ void kv_outer_kernel(const ushort_t* __restrict__ qkv,  // [T][3072] bf16
                                float* __restrict__ G) {           // [16][64][64][64]
  const int c = blockIdx.x, h = blockIdx.y;
  __shared__ __align__(16) ushort_t Ksh[64][64];  // [d1][t]  (A = K^T)
  __shared__ __align__(16) ushort_t Vsh[64][64];  // [d2][t]  (B layout [n][k])
  const int tid = threadIdx.x;
  const int lane = tid & 31, w = tid >> 5;
  const int wm = w >> 1, wn = w & 1;
  const int g = lane >> 4, r16 = lane & 15;

#pragma unroll
  for (int j = 0; j < 2; ++j) {
    int idx = tid + j * 256;              // 0..511
    int row = idx >> 3;                   // t within chunk
    int cg = (idx & 7) * 8;               // d group
    const size_t base = (size_t)(c * 64 + row) * 3072 + h * 64 + cg;
    uint4 kd = *(const uint4*)(qkv + base + 1024);
    uint4 vd = *(const uint4*)(qkv + base + 2048);
    const ushort_t* ks = (const ushort_t*)&kd;
    const ushort_t* vs = (const ushort_t*)&vd;
#pragma unroll
    for (int i = 0; i < 8; ++i) { Ksh[cg + i][row] = ks[i]; Vsh[cg + i][row] = vs[i]; }
  }
  __syncthreads();

  v8f c0 = {}, c1 = {};
#pragma unroll
  for (int k0 = 0; k0 < 64; k0 += 32) {
    FragBF a, b0, b1;
#pragma unroll
    for (int v = 0; v < 8; ++v) {
      a.u[v]  = *(const uint_t*)&Ksh[wm * 16 + r16][a_kidx(v, g) + k0];
      b0.u[v] = *(const uint_t*)&Vsh[wn * 32 + r16][b_kidx(v, g) + k0];
      b1.u[v] = *(const uint_t*)&Vsh[wn * 32 + 16 + r16][b_kidx(v, g) + k0];
    }
    c0 = wmma_bf16(a, b0, c0);
    c1 = wmma_bf16(a, b1, c1);
  }

  float* Gp = G + ((size_t)h * 64 + c) * 4096;
#pragma unroll
  for (int r = 0; r < 8; ++r) {
    int m = wm * 16 + 8 * g + r, n = wn * 32 + r16;
    Gp[m * 64 + n]      = c0[r];
    Gp[m * 64 + n + 16] = c1[r];
  }
}

// ---------------------------------------------------------------- Pass B: exclusive prefix
__global__ void prefix_kernel(const float* __restrict__ state,  // [16][64][64]
                              const float* __restrict__ G,      // [16][64][64][64]
                              ushort_t* __restrict__ Sbf,       // [16][64][64][64]
                              float* __restrict__ new_state) {  // [16][64][64]
  const int h = blockIdx.x, tid = threadIdx.x;
  float acc[16];
  const float* st = state + (size_t)h * 4096 + tid * 16;
#pragma unroll
  for (int i = 0; i < 16; ++i) acc[i] = st[i];
  for (int c = 0; c < 64; ++c) {
    ushort_t* sb = Sbf + ((size_t)h * 64 + c) * 4096 + tid * 16;
    const float* gp = G + ((size_t)h * 64 + c) * 4096 + tid * 16;
#pragma unroll
    for (int i = 0; i < 16; ++i) sb[i] = f32_to_bf16(acc[i]);  // S_c excludes G_c
#pragma unroll
    for (int i = 0; i < 16; ++i) acc[i] += gp[i];
  }
  float* ns = new_state + (size_t)h * 4096 + tid * 16;
#pragma unroll
  for (int i = 0; i < 16; ++i) ns[i] = acc[i];
}

// ---------------------------------------------------------------- Pass C:
// out_c = Q_c @ S_c + tril(Q_c K_c^T) @ V_c ; grid (64 chunks, 16 heads).
__global__ void attn_chunk_kernel(const ushort_t* __restrict__ qkv,
                                  const ushort_t* __restrict__ Sbf,  // [16][64][64][64]
                                  ushort_t* __restrict__ outb) {     // [T][1024] bf16
  const int c = blockIdx.x, h = blockIdx.y;
  __shared__ __align__(16) ushort_t Qs[64][64];   // [t][d]    A layout
  __shared__ __align__(16) ushort_t Kb[64][64];   // [t2][d]   B layout for Q K^T
  __shared__ __align__(16) ushort_t Ssh[64][64];  // [d2][d]   B layout for Q S
  __shared__ __align__(16) ushort_t Vsh[64][64];  // [d2][t2]  B layout for P V
  __shared__ __align__(16) ushort_t Ps[64][64];   // [t][t2]   masked P (A layout)
  const int tid = threadIdx.x;
  const int lane = tid & 31, w = tid >> 5;
  const int wm = w >> 1, wn = w & 1;
  const int g = lane >> 4, r16 = lane & 15;

#pragma unroll
  for (int j = 0; j < 2; ++j) {
    int idx = tid + j * 256;
    int row = idx >> 3;
    int cg = (idx & 7) * 8;
    const size_t base = (size_t)(c * 64 + row) * 3072 + h * 64 + cg;
    uint4 qd = *(const uint4*)(qkv + base);
    uint4 kd = *(const uint4*)(qkv + base + 1024);
    uint4 vd = *(const uint4*)(qkv + base + 2048);
    uint4 sd = *(const uint4*)(Sbf + ((size_t)h * 64 + c) * 4096 + row * 64 + cg);
    *(uint4*)&Qs[row][cg] = qd;
    *(uint4*)&Kb[row][cg] = kd;
    const ushort_t* vs = (const ushort_t*)&vd;
    const ushort_t* ss = (const ushort_t*)&sd;
#pragma unroll
    for (int i = 0; i < 8; ++i) { Vsh[cg + i][row] = vs[i]; Ssh[cg + i][row] = ss[i]; }
  }
  __syncthreads();

  v8f p0 = {}, p1 = {}, o0 = {}, o1 = {};
#pragma unroll
  for (int k0 = 0; k0 < 64; k0 += 32) {
    FragBF a, bk0, bk1, bs0, bs1;
#pragma unroll
    for (int v = 0; v < 8; ++v) {
      a.u[v]   = *(const uint_t*)&Qs[wm * 16 + r16][a_kidx(v, g) + k0];
      bk0.u[v] = *(const uint_t*)&Kb[wn * 32 + r16][b_kidx(v, g) + k0];
      bk1.u[v] = *(const uint_t*)&Kb[wn * 32 + 16 + r16][b_kidx(v, g) + k0];
      bs0.u[v] = *(const uint_t*)&Ssh[wn * 32 + r16][b_kidx(v, g) + k0];
      bs1.u[v] = *(const uint_t*)&Ssh[wn * 32 + 16 + r16][b_kidx(v, g) + k0];
    }
    p0 = wmma_bf16(a, bk0, p0);
    p1 = wmma_bf16(a, bk1, p1);
    o0 = wmma_bf16(a, bs0, o0);
    o1 = wmma_bf16(a, bs1, o1);
  }

  // causal mask within diagonal chunk (row >= col kept), write P in A layout
#pragma unroll
  for (int r = 0; r < 8; ++r) {
    int m = wm * 16 + 8 * g + r;
    int n0c = wn * 32 + r16, n1c = n0c + 16;
    Ps[m][n0c] = f32_to_bf16(m >= n0c ? p0[r] : 0.f);
    Ps[m][n1c] = f32_to_bf16(m >= n1c ? p1[r] : 0.f);
  }
  __syncthreads();

#pragma unroll
  for (int k0 = 0; k0 < 64; k0 += 32) {
    FragBF a, b0, b1;
#pragma unroll
    for (int v = 0; v < 8; ++v) {
      a.u[v]  = *(const uint_t*)&Ps[wm * 16 + r16][a_kidx(v, g) + k0];
      b0.u[v] = *(const uint_t*)&Vsh[wn * 32 + r16][b_kidx(v, g) + k0];
      b1.u[v] = *(const uint_t*)&Vsh[wn * 32 + 16 + r16][b_kidx(v, g) + k0];
    }
    o0 = wmma_bf16(a, b0, o0);
    o1 = wmma_bf16(a, b1, o1);
  }

#pragma unroll
  for (int r = 0; r < 8; ++r) {
    int m = wm * 16 + 8 * g + r;
    int n = wn * 32 + r16;
    outb[(size_t)(c * 64 + m) * 1024 + h * 64 + n]      = f32_to_bf16(o0[r]);
    outb[(size_t)(c * 64 + m) * 1024 + h * 64 + n + 16] = f32_to_bf16(o1[r]);
  }
}

// ---------------------------------------------------------------- host launcher
extern "C" void kernel_launch(void* const* d_in, const int* in_sizes, int n_in,
                              void* d_out, int out_size, void* d_ws, size_t ws_size,
                              hipStream_t stream) {
  const float* state = (const float*)d_in[0];   // 16*64*64
  const float* x     = (const float*)d_in[1];   // 4096*1024
  const float* w_qkv = (const float*)d_in[2];   // 1024*3072
  const float* b_qkv = (const float*)d_in[3];   // 3072
  const float* w_out = (const float*)d_in[4];   // 1024*1024
  const float* b_out = (const float*)d_in[5];   // 1024

  float* out_state = (float*)d_out;             // 65536 floats
  float* out_y     = (float*)d_out + 65536;     // 4096*1024 floats

  char* ws = (char*)d_ws;
  ushort_t* x_bf     = (ushort_t*)(ws + 0);                    //  8 MB [4096][1024]
  ushort_t* w_qkv_bt = (ushort_t*)(ws + 8388608);              //  6 MB [3072][1024] (=W^T)
  ushort_t* w_out_bt = (ushort_t*)(ws + 14680064);             //  2 MB [1024][1024] (=W^T)
  ushort_t* qkv_bf   = (ushort_t*)(ws + 16777216);             // 24 MB [4096][3072]
  float*    G        = (float*)   (ws + 41943040);             // 16 MB [16][64][64][64]
  ushort_t* S_bf     = (ushort_t*)(ws + 58720256);             //  8 MB [16][64][64][64]
  ushort_t* out_bf   = (ushort_t*)(ws + 67108864);             //  8 MB [4096][1024]

  // 1) casts to bf16 (weights transposed so GEMM B-tiles are K-contiguous)
  cast_f32_to_bf16<<<(4194304 + 255) / 256, 256, 0, stream>>>(x, x_bf, 4194304);
  transpose_cast_f32_bf16<<<dim3(3072 / 32, 1024 / 32), dim3(32, 8), 0, stream>>>(
      w_qkv, w_qkv_bt, 1024, 3072);
  transpose_cast_f32_bf16<<<dim3(1024 / 32, 1024 / 32), dim3(32, 8), 0, stream>>>(
      w_out, w_out_bt, 1024, 1024);

  // 2) qkv = x @ w_qkv + b_qkv   (4096 x 3072, bf16 out)
  gemm_bf16_wmma<true><<<dim3(3072 / 128, 4096 / 128), 256, 0, stream>>>(
      x_bf, w_qkv_bt, b_qkv, (void*)qkv_bf, 4096, 3072, 1024);

  // 3) per-chunk G_c = K_c^T V_c
  kv_outer_kernel<<<dim3(64, 16), 256, 0, stream>>>(qkv_bf, G);

  // 4) exclusive prefix over chunks -> S_c (bf16) and new_state (fp32, output 0)
  prefix_kernel<<<16, 256, 0, stream>>>(state, G, S_bf, out_state);

  // 5) out_c = Q_c S_c + tril(Q_c K_c^T) V_c
  attn_chunk_kernel<<<dim3(64, 16), 256, 0, stream>>>(qkv_bf, S_bf, out_bf);

  // 6) y = out @ w_out + b_out   (4096 x 1024, fp32 out -> output 1)
  gemm_bf16_wmma<false><<<dim3(1024 / 128, 4096 / 128), 256, 0, stream>>>(
      out_bf, w_out_bt, b_out, (void*)out_y, 4096, 1024, 1024);
}
